// VGAE_44633300140358
// MI455X (gfx1250) — matrix operand order repaired and checked by reading, not compiled, not run on hardware
//
#include <hip/hip_runtime.h>

#define IN_DIM 512
#define H1 32
#define H2 16

typedef __attribute__((ext_vector_type(16))) __bf16 v16bf;
typedef __attribute__((ext_vector_type(8)))  float  v8f;

// ---------------------------------------------------------------------------
// Zero the accumulation workspace (atomic-add targets must start at 0).
// float4 stores; total count is N*128 floats (divisible by 4).
// ---------------------------------------------------------------------------
__global__ __launch_bounds__(256) void vgae_zero4(float4* __restrict__ p, long long n4) {
  long long i = (long long)blockIdx.x * blockDim.x + threadIdx.x;
  if (i < n4) p[i] = make_float4(0.f, 0.f, 0.f, 0.f);
}

// ---------------------------------------------------------------------------
// GEMM1: dense1 = x @ W1 + b1   ([N,512] x [512,32] -> [N,32])
// bf16 WMMA with f32 accumulate. W1 staged transposed in LDS (padded stride).
// Each wave computes a 16x32 output tile; 16 K-steps x 2 N-tiles = 32 WMMAs.
// ---------------------------------------------------------------------------
__global__ __launch_bounds__(256) void vgae_gemm1(
    const float* __restrict__ x, const float* __restrict__ W1,
    const float* __restrict__ b1, float* __restrict__ out, int n)
{
  const int WTS = IN_DIM + 8;                 // pad to avoid LDS bank conflicts
  __shared__ __bf16 Wt[H1 * (IN_DIM + 8)];    // Wt[col][k], bf16
  for (int i = threadIdx.x; i < IN_DIM * H1; i += blockDim.x) {
    int k = i >> 5, c = i & 31;               // W1 row-major [512][32]
    Wt[c * WTS + k] = (__bf16)W1[i];
  }
  __syncthreads();

  const int wave = threadIdx.x >> 5;
  const int lane = threadIdx.x & 31;
  const int rowBase = (blockIdx.x * 8 + wave) * 16;
  if (rowBase >= n) return;                   // wave-uniform: EXEC stays all-1s

  const int half = lane >> 4;                 // 0: lanes 0-15, 1: lanes 16-31
  const int col  = lane & 15;
  const int rowA = min(rowBase + col, n - 1); // clamp tail rows for loads

  v8f acc0, acc1;
  const float bias0 = b1[col];
  const float bias1 = b1[16 + col];
  #pragma unroll
  for (int i = 0; i < 8; ++i) { acc0[i] = bias0; acc1[i] = bias1; }

  const float* xrow = x + (size_t)rowA * IN_DIM;
  for (int kb = 0; kb < IN_DIM; kb += 32) {
    // A fragment: lane covers K in [16*half, 16*half+16)
    v16bf a;
    const float* xp = xrow + kb + half * 16;
    #pragma unroll
    for (int t = 0; t < 16; ++t) {
      a[t]     = (__bf16)xp[t];
    }
    // B fragments: lane's column, 16 contiguous K-values from LDS
    v16bf b0, b1v;
    const __bf16* wp0 = &Wt[col        * WTS + kb + 16 * half];
    const __bf16* wp1 = &Wt[(16 + col) * WTS + kb + 16 * half];
    #pragma unroll
    for (int t = 0; t < 16; ++t) { b0[t] = wp0[t]; b1v[t] = wp1[t]; }

    acc0 = __builtin_amdgcn_wmma_f32_16x16x32_bf16(
        false, a, false, b0, (short)0, acc0, false, false);
    acc1 = __builtin_amdgcn_wmma_f32_16x16x32_bf16(
        false, a, false, b1v, (short)0, acc1, false, false);
  }

  const int m0 = rowBase + half * 8;          // C/D layout: VGPR r -> row m0+r
  if (rowBase + 16 <= n) {
    // fast path: whole tile in range, no per-element predication
    #pragma unroll
    for (int i = 0; i < 8; ++i) {
      int row = m0 + i;
      out[(size_t)row * H1 + col]      = acc0[i];
      out[(size_t)row * H1 + 16 + col] = acc1[i];
    }
  } else {
    #pragma unroll
    for (int i = 0; i < 8; ++i) {
      int row = m0 + i;
      if (row < n) {
        out[(size_t)row * H1 + col]      = acc0[i];
        out[(size_t)row * H1 + 16 + col] = acc1[i];
      }
    }
  }
}

// ---------------------------------------------------------------------------
// SpMM pass 1: h1agg[r] += val[e] * dense1[c]  (F = 32)
// One wave per edge: edge scalars via uniform (s_load) reads, 32 lanes cover
// the feature row -> one coalesced 128B gather + one coalesced atomic scatter.
// ---------------------------------------------------------------------------
__global__ __launch_bounds__(256) void vgae_spmm32(
    const int* __restrict__ erow, const int* __restrict__ ecol,
    const float* __restrict__ eval, const float* __restrict__ dense,
    float* __restrict__ out, long long E)
{
  const int lane = threadIdx.x & 31;
  const int wave = __builtin_amdgcn_readfirstlane(threadIdx.x >> 5);
  const long long e = (long long)blockIdx.x * 8 + wave;   // wave-uniform
  if (e >= E) return;
  const float v = eval[e];                                // scalar loads
  const int   c = ecol[e];
  const int   r = erow[e];
  atomicAdd(&out[(size_t)r * H1 + lane], v * dense[(size_t)c * H1 + lane]);
}

// ---------------------------------------------------------------------------
// Dense2: mu_d = relu(h1) @ Wmu + bmu ; ls_d = relu(h1) @ Wls + bls
// K=32 == one WMMA per 16-row tile per head. ReLU fused into bf16 convert.
// ---------------------------------------------------------------------------
__global__ __launch_bounds__(256) void vgae_dense2(
    const float* __restrict__ h1, const float* __restrict__ Wmu,
    const float* __restrict__ bmu, const float* __restrict__ Wls,
    const float* __restrict__ bls, float* __restrict__ mu_d,
    float* __restrict__ ls_d, int n)
{
  const int WS = H1 + 8;
  __shared__ __bf16 WmT[H2 * (H1 + 8)];
  __shared__ __bf16 WlT[H2 * (H1 + 8)];
  for (int i = threadIdx.x; i < H1 * H2; i += blockDim.x) {
    int k = i >> 4, c = i & 15;               // W row-major [32][16]
    WmT[c * WS + k] = (__bf16)Wmu[i];
    WlT[c * WS + k] = (__bf16)Wls[i];
  }
  __syncthreads();

  const int wave = threadIdx.x >> 5;
  const int lane = threadIdx.x & 31;
  const int rowBase = (blockIdx.x * 8 + wave) * 16;
  if (rowBase >= n) return;

  const int half = lane >> 4;
  const int col  = lane & 15;
  const int rowA = min(rowBase + col, n - 1);

  const float* hp = h1 + (size_t)rowA * H1 + half * 16;
  v16bf a;
  #pragma unroll
  for (int t = 0; t < 16; ++t) {
    a[t]     = (__bf16)fmaxf(hp[t], 0.0f);       // fused ReLU
  }

  v16bf bm, bl;
  const __bf16* pm = &WmT[col * WS + 16 * half];
  const __bf16* pl = &WlT[col * WS + 16 * half];
  #pragma unroll
  for (int t = 0; t < 16; ++t) { bm[t] = pm[t]; bl[t] = pl[t]; }

  v8f cm, cl;
  const float biasM = bmu[col];
  const float biasL = bls[col];
  #pragma unroll
  for (int i = 0; i < 8; ++i) { cm[i] = biasM; cl[i] = biasL; }

  cm = __builtin_amdgcn_wmma_f32_16x16x32_bf16(
      false, a, false, bm, (short)0, cm, false, false);
  cl = __builtin_amdgcn_wmma_f32_16x16x32_bf16(
      false, a, false, bl, (short)0, cl, false, false);

  const int m0 = rowBase + half * 8;
  if (rowBase + 16 <= n) {
    #pragma unroll
    for (int i = 0; i < 8; ++i) {
      int row = m0 + i;
      mu_d[(size_t)row * H2 + col] = cm[i];
      ls_d[(size_t)row * H2 + col] = cl[i];
    }
  } else {
    #pragma unroll
    for (int i = 0; i < 8; ++i) {
      int row = m0 + i;
      if (row < n) {
        mu_d[(size_t)row * H2 + col] = cm[i];
        ls_d[(size_t)row * H2 + col] = cl[i];
      }
    }
  }
}

// ---------------------------------------------------------------------------
// SpMM pass 2: mu_o and ls_o aggregated in one pass.
// One wave per edge: lanes 0-15 -> mu features, lanes 16-31 -> logstd features.
// ---------------------------------------------------------------------------
__global__ __launch_bounds__(256) void vgae_spmm16x2(
    const int* __restrict__ erow, const int* __restrict__ ecol,
    const float* __restrict__ eval,
    const float* __restrict__ mu_d, const float* __restrict__ ls_d,
    float* __restrict__ mu_o, float* __restrict__ ls_o, long long E)
{
  const int lane = threadIdx.x & 31;
  const int wave = __builtin_amdgcn_readfirstlane(threadIdx.x >> 5);
  const long long e = (long long)blockIdx.x * 8 + wave;   // wave-uniform
  if (e >= E) return;
  const float v = eval[e];                                // scalar loads
  const int   c = ecol[e];
  const int   r = erow[e];
  const int   f = lane & 15;
  const float* __restrict__ src = (lane < 16) ? mu_d : ls_d;
  float* __restrict__       dst = (lane < 16) ? mu_o : ls_o;
  atomicAdd(&dst[(size_t)r * H2 + f], v * src[(size_t)c * H2 + f]);
}

// ---------------------------------------------------------------------------
// Combine: z = mu + eps*exp(logstd); write tuple (z, mu, logstd) concatenated.
// float4 per thread (n16 = N*16 is divisible by 4).
// ---------------------------------------------------------------------------
__global__ __launch_bounds__(256) void vgae_combine4(
    const float4* __restrict__ mu, const float4* __restrict__ ls,
    const float4* __restrict__ eps, float* __restrict__ out, int n16)
{
  int i = blockIdx.x * blockDim.x + threadIdx.x;
  if (i >= (n16 >> 2)) return;
  float4 m = mu[i];
  float4 l = ls[i];
  float4 ep = eps[i];
  float4 z;
  z.x = m.x + ep.x * __expf(l.x);
  z.y = m.y + ep.y * __expf(l.y);
  z.z = m.z + ep.z * __expf(l.z);
  z.w = m.w + ep.w * __expf(l.w);
  ((float4*)out)[i]                 = z;   // z
  ((float4*)(out + n16))[i]         = m;   // mu
  ((float4*)(out + 2 * n16))[i]     = l;   // logstd
}

// ---------------------------------------------------------------------------
extern "C" void kernel_launch(void* const* d_in, const int* in_sizes, int n_in,
                              void* d_out, int out_size, void* d_ws, size_t ws_size,
                              hipStream_t stream)
{
  const float* x    = (const float*)d_in[0];
  const float* W1   = (const float*)d_in[1];
  const float* b1   = (const float*)d_in[2];
  const float* Wmu  = (const float*)d_in[3];
  const float* bmu  = (const float*)d_in[4];
  const float* Wls  = (const float*)d_in[5];
  const float* bls  = (const float*)d_in[6];
  const float* eps  = (const float*)d_in[7];
  const float* eval = (const float*)d_in[8];
  const int*   erow = (const int*)d_in[9];
  const int*   ecol = (const int*)d_in[10];

  const int       n = in_sizes[7] / H2;     // N from eps [N,16]
  const long long e = in_sizes[8];          // E from edge_val

  float* ws     = (float*)d_ws;             // needs N*128 floats (~51 MB)
  float* dense1 = ws;                       // [N,32]
  float* h1agg  = ws + (size_t)n * 32;      // [N,32]  (atomic target)
  float* mu_d   = ws + (size_t)n * 64;      // [N,16]
  float* ls_d   = ws + (size_t)n * 80;      // [N,16]
  float* mu_o   = ws + (size_t)n * 96;      // [N,16]  (atomic target)
  float* ls_o   = ws + (size_t)n * 112;     // [N,16]  (atomic target)

  {
    long long n4 = (long long)n * 32;       // N*128 floats / 4
    unsigned g = (unsigned)((n4 + 255) / 256);
    vgae_zero4<<<g, 256, 0, stream>>>((float4*)ws, n4);
  }
  {
    int blocks = (n + 127) / 128;           // 8 waves x 16 rows per block
    vgae_gemm1<<<blocks, 256, 0, stream>>>(x, W1, b1, dense1, n);
  }
  {
    unsigned g = (unsigned)((e + 7) / 8);   // one wave per edge, 8 waves/block
    vgae_spmm32<<<g, 256, 0, stream>>>(erow, ecol, eval, dense1, h1agg, e);
  }
  {
    int blocks = (n + 127) / 128;
    vgae_dense2<<<blocks, 256, 0, stream>>>(h1agg, Wmu, bmu, Wls, bls,
                                            mu_d, ls_d, n);
  }
  {
    unsigned g = (unsigned)((e + 7) / 8);
    vgae_spmm16x2<<<g, 256, 0, stream>>>(erow, ecol, eval, mu_d, ls_d,
                                         mu_o, ls_o, e);
  }
  {
    int n16 = n * H2;
    unsigned g = (unsigned)(((n16 >> 2) + 255) / 256);
    vgae_combine4<<<g, 256, 0, stream>>>((const float4*)mu_o,
                                         (const float4*)ls_o,
                                         (const float4*)eps,
                                         (float*)d_out, n16);
  }
}
